// DMPNNEncoder_32349693673952
// MI455X (gfx1250) — matrix operand order repaired and checked by reading, not compiled, not run on hardware
//
#include <hip/hip_runtime.h>
#include <stddef.h>

// ---- problem constants (from reference) ----
#define EEDGES   200000
#define NATOMS   50000
#define HID      300
#define HP       320      // hidden padded to 20 WMMA n-tiles
#define NTILES   20
#define ADIM     133
#define BDIM     14
#define K0R      147      // BDIM+ADIM
#define K0P      160      // padded K for h0 gemm (5 k-steps)
#define KT0      5
#define KMP      320      // padded K for message gemm (10 k-steps)
#define KTM      10
#define KAR      433      // ADIM+HID
#define KAP      448      // padded K for atom gemm (14 k-steps)
#define KTA      14
#define NMOL     2048
#define FDIM     200
#define OUTW     500      // HID + FDIM

typedef __attribute__((ext_vector_type(16))) __bf16 v16bf;
typedef __attribute__((ext_vector_type(8)))  float  v8f;

union ABu { v16bf v; uint4 q[2]; unsigned short h[16]; };

// ---- bf16 helpers (round-to-nearest-even) ----
__device__ __forceinline__ unsigned short f2b(float f) {
  union { float f; unsigned u; } x; x.f = f;
  unsigned r = x.u + 0x7FFFu + ((x.u >> 16) & 1u);
  return (unsigned short)(r >> 16);
}
__device__ __forceinline__ float b2f(unsigned short b) {
  union { unsigned u; float f; } x; x.u = ((unsigned)b) << 16;
  return x.f;
}
__device__ __forceinline__ void atomAddF(float* p, float v) {
  __hip_atomic_fetch_add(p, v, __ATOMIC_RELAXED, __HIP_MEMORY_SCOPE_AGENT);
}

// A fragment (16x32 bf16) from LDS-staged row-major [rows][KP] tile.
// ISA layout: lane<16 -> row=lane, K chunks {0..7,16..23}; lane>=16 -> row=lane-16,
// chunks {8..15,24..31}. Two contiguous 16B loads per lane.
__device__ __forceinline__ v16bf frag_a(const unsigned short* base, int KP,
                                        int rowbase, int kbase, int lane) {
  int row = rowbase + (lane & 15);
  const unsigned short* p = base + (size_t)row * KP + kbase + ((lane & 16) ? 8 : 0);
  ABu u;
  u.q[0] = *(const uint4*)p;
  u.q[1] = *(const uint4*)(p + 16);
  return u.v;
}

// B fragment (32x16 bf16) from pre-packed weights: tile (kt,nt), each lane owns
// 16 contiguous halfwords (its column's 16 K values).
__device__ __forceinline__ v16bf frag_b(const unsigned short* wpk, int kt, int nt, int lane) {
  const unsigned short* p = wpk + (((size_t)(kt * NTILES + nt)) * 32 + lane) * 16;
  ABu u;
  u.q[0] = *(const uint4*)p;
  u.q[1] = *(const uint4*)(p + 8);
  return u.v;
}

// ---- weight packing: W [Kreal][300] f32 row-major -> WMMA-B tiled bf16 ----
__global__ __launch_bounds__(256) void k_pack_w(const float* __restrict__ W, int Kreal,
                                                int KT, unsigned short* __restrict__ out) {
  int idx = blockIdx.x * 256 + threadIdx.x;
  int total = KT * NTILES * 512;
  if (idx >= total) return;
  int t  = idx >> 9;          // tile
  int r2 = idx & 511;         // entry in tile
  int lane = r2 >> 4, j = r2 & 15;
  int kt = t / NTILES, nt = t % NTILES;
  int k = kt * 32 + ((j < 8) ? ((lane >> 4) * 8 + j) : (16 + (lane >> 4) * 8 + (j - 8)));
  int n = nt * 16 + (lane & 15);
  float v = (k < Kreal && n < HID) ? W[(size_t)k * HID + n] : 0.0f;
  out[idx] = f2b(v);
}

__global__ __launch_bounds__(256) void k_pack_bias(const float* __restrict__ b,
                                                   float* __restrict__ bp) {
  int i = blockIdx.x * 256 + threadIdx.x;
  if (i < HP) bp[i] = (i < HID) ? b[i] : 0.0f;
}

// ---- h0 = relu(concat(edge_attr, x[src]) @ Wi + bi), bf16 out [E][HP] ----
__global__ __launch_bounds__(256) void k_h0(const float* __restrict__ x,
                                            const float* __restrict__ ea,
                                            const int* __restrict__ esrc,
                                            const unsigned short* __restrict__ wi_pk,
                                            const float* __restrict__ bi_p,
                                            unsigned short* __restrict__ h0) {
  __shared__ unsigned short As[64 * K0P];
  __shared__ int srcs[64];
  int tid = threadIdx.x;
  int e0 = blockIdx.x * 64;
  if (tid < 64) srcs[tid] = esrc[e0 + tid];
  __syncthreads();
  for (int idx = tid; idx < 64 * K0P; idx += 256) {
    int r = idx / K0P, k = idx - r * K0P;
    float v = 0.0f;
    if (k < BDIM)       v = ea[(size_t)(e0 + r) * BDIM + k];
    else if (k < K0R)   v = x[(size_t)srcs[r] * ADIM + (k - BDIM)];
    As[idx] = f2b(v);
  }
  __syncthreads();

  int wave = tid >> 5, lane = tid & 31;
  int mt = wave >> 1, nt0 = (wave & 1) * 10;
  v8f zacc = {0.f,0.f,0.f,0.f,0.f,0.f,0.f,0.f};
  v8f acc[10];
#pragma unroll
  for (int i = 0; i < 10; ++i) acc[i] = zacc;

#pragma unroll
  for (int kt = 0; kt < KT0; ++kt) {
    v16bf a = frag_a(As, K0P, mt * 16, kt * 32, lane);
#pragma unroll
    for (int i = 0; i < 10; ++i) {
      v16bf b = frag_b(wi_pk, kt, nt0 + i, lane);
      acc[i] = __builtin_amdgcn_wmma_f32_16x16x32_bf16(false, a, false, b,
                                                       (short)0, acc[i], false, false);
    }
  }
  int mo = (lane >> 4) * 8;
#pragma unroll
  for (int i = 0; i < 10; ++i) {
    int col = (nt0 + i) * 16 + (lane & 15);
    float bias = bi_p[col];
#pragma unroll
    for (int r = 0; r < 8; ++r) {
      int g = e0 + mt * 16 + mo + r;
      float v = acc[i][r] + bias;
      h0[(size_t)g * HP + col] = f2b(v > 0.f ? v : 0.f);
    }
  }
}

// ---- zero an f32 buffer ----
__global__ __launch_bounds__(256) void k_zero(float* __restrict__ p, size_t n) {
  size_t i = (size_t)blockIdx.x * 256 + threadIdx.x;
  size_t stride = (size_t)gridDim.x * 256;
  for (; i < n; i += stride) p[i] = 0.0f;
}

// ---- segment scatter-add: acc[idx[e]][k] += h[e][k] (bf16 src, f32 dst) ----
__global__ __launch_bounds__(256) void k_scatter(const unsigned short* __restrict__ h,
                                                 const int* __restrict__ idxarr,
                                                 float* __restrict__ acc) {
  size_t idx = (size_t)blockIdx.x * 256 + threadIdx.x;
  const size_t total = (size_t)EEDGES * (HP / 2);
  if (idx >= total) return;
  int e = (int)(idx / (HP / 2));
  int j = (int)(idx - (size_t)e * (HP / 2));
  unsigned d = *(const unsigned*)(h + (size_t)e * HP + 2 * j);
  float f0 = b2f((unsigned short)(d & 0xFFFFu));
  float f1 = b2f((unsigned short)(d >> 16));
  int a = idxarr[e];
  float* base = acc + (size_t)a * HP + 2 * j;
  if (f0 != 0.0f) atomAddF(base, f0);       // post-ReLU: ~half are exact zeros
  if (f1 != 0.0f) atomAddF(base + 1, f1);
}

// ---- message pass: h_next = relu(h0 + (inc[src]-h[rev]) @ Wm + bm) ----
__global__ __launch_bounds__(256) void k_mp(const int* __restrict__ esrc,
                                            const int* __restrict__ b2rev,
                                            const float* __restrict__ inc,
                                            const unsigned short* __restrict__ hcur,
                                            const unsigned short* __restrict__ h0,
                                            const unsigned short* __restrict__ wm_pk,
                                            const float* __restrict__ bm_p,
                                            unsigned short* __restrict__ hnxt) {
  __shared__ unsigned short As[64 * KMP];
  __shared__ int srcs[64];
  __shared__ int revs[64];
  int tid = threadIdx.x;
  int e0 = blockIdx.x * 64;
  if (tid < 64)      srcs[tid] = esrc[e0 + tid];
  else if (tid < 128) revs[tid - 64] = b2rev[e0 + tid - 64];
  __syncthreads();
  for (int idx = tid; idx < 64 * KMP; idx += 256) {
    int r = idx / KMP, k = idx - r * KMP;
    float v = 0.0f;
    if (k < HID)
      v = inc[(size_t)srcs[r] * HP + k] - b2f(hcur[(size_t)revs[r] * HP + k]);
    As[idx] = f2b(v);
  }
  __syncthreads();

  int wave = tid >> 5, lane = tid & 31;
  int mt = wave >> 1, nt0 = (wave & 1) * 10;
  v8f zacc = {0.f,0.f,0.f,0.f,0.f,0.f,0.f,0.f};
  v8f acc[10];
#pragma unroll
  for (int i = 0; i < 10; ++i) acc[i] = zacc;

  for (int kt = 0; kt < KTM; ++kt) {
    v16bf a = frag_a(As, KMP, mt * 16, kt * 32, lane);
#pragma unroll
    for (int i = 0; i < 10; ++i) {
      v16bf b = frag_b(wm_pk, kt, nt0 + i, lane);
      acc[i] = __builtin_amdgcn_wmma_f32_16x16x32_bf16(false, a, false, b,
                                                       (short)0, acc[i], false, false);
    }
  }
  int mo = (lane >> 4) * 8;
#pragma unroll
  for (int i = 0; i < 10; ++i) {
    int col = (nt0 + i) * 16 + (lane & 15);
    float bias = bm_p[col];
#pragma unroll
    for (int r = 0; r < 8; ++r) {
      int g = e0 + mt * 16 + mo + r;
      size_t o = (size_t)g * HP + col;
      float v = acc[i][r] + b2f(h0[o]) + bias;
      hnxt[o] = f2b(v > 0.f ? v : 0.f);
    }
  }
}

// ---- h_v = relu(concat(x, m_v) @ Wa + ba), f32 out [NATOMS][HP] ----
__global__ __launch_bounds__(256) void k_hv(const float* __restrict__ x,
                                            const float* __restrict__ mv,
                                            const unsigned short* __restrict__ wa_pk,
                                            const float* __restrict__ ba_p,
                                            float* __restrict__ hv) {
  __shared__ unsigned short As[64 * KAP];
  int tid = threadIdx.x;
  int a0 = blockIdx.x * 64;
  for (int idx = tid; idx < 64 * KAP; idx += 256) {
    int r = idx / KAP, k = idx - r * KAP;
    int at = a0 + r;
    float v = 0.0f;
    if (at < NATOMS) {
      if (k < ADIM)      v = x[(size_t)at * ADIM + k];
      else if (k < KAR)  v = mv[(size_t)at * HP + (k - ADIM)];
    }
    As[idx] = f2b(v);
  }
  __syncthreads();

  int wave = tid >> 5, lane = tid & 31;
  int mt = wave >> 1, nt0 = (wave & 1) * 10;
  v8f zacc = {0.f,0.f,0.f,0.f,0.f,0.f,0.f,0.f};
  v8f acc[10];
#pragma unroll
  for (int i = 0; i < 10; ++i) acc[i] = zacc;

  for (int kt = 0; kt < KTA; ++kt) {
    v16bf a = frag_a(As, KAP, mt * 16, kt * 32, lane);
#pragma unroll
    for (int i = 0; i < 10; ++i) {
      v16bf b = frag_b(wa_pk, kt, nt0 + i, lane);
      acc[i] = __builtin_amdgcn_wmma_f32_16x16x32_bf16(false, a, false, b,
                                                       (short)0, acc[i], false, false);
    }
  }
  int mo = (lane >> 4) * 8;
#pragma unroll
  for (int i = 0; i < 10; ++i) {
    int col = (nt0 + i) * 16 + (lane & 15);
    float bias = ba_p[col];
#pragma unroll
    for (int r = 0; r < 8; ++r) {
      int g = a0 + mt * 16 + mo + r;
      if (g < NATOMS) {
        float v = acc[i][r] + bias;
        hv[(size_t)g * HP + col] = v > 0.f ? v : 0.f;
      }
    }
  }
}

// ---- out init: hm part = 0, features part copied ----
__global__ __launch_bounds__(256) void k_outinit(float* __restrict__ out,
                                                 const float* __restrict__ feats) {
  int idx = blockIdx.x * 256 + threadIdx.x;
  if (idx >= NMOL * OUTW) return;
  int m = idx / OUTW, c = idx - m * OUTW;
  out[idx] = (c < HID) ? 0.0f : feats[(size_t)m * FDIM + (c - HID)];
}

// ---- per-molecule readout: out[mol[a]][k] += h_v[a][k] ----
__global__ __launch_bounds__(256) void k_mol(const float* __restrict__ hv,
                                             const int* __restrict__ mol,
                                             float* __restrict__ out) {
  size_t idx = (size_t)blockIdx.x * 256 + threadIdx.x;
  const size_t total = (size_t)NATOMS * HID;
  if (idx >= total) return;
  int a = (int)(idx / HID);
  int k = (int)(idx - (size_t)a * HID);
  float v = hv[(size_t)a * HP + k];
  if (v != 0.0f) atomAddF(out + (size_t)mol[a] * OUTW + k, v);
}

extern "C" void kernel_launch(void* const* d_in, const int* in_sizes, int n_in,
                              void* d_out, int out_size, void* d_ws, size_t ws_size,
                              hipStream_t stream) {
  const float* x    = (const float*)d_in[0];
  const float* ea   = (const float*)d_in[1];
  const int*  esrc  = (const int*)d_in[2];
  const int*  edst  = (const int*)d_in[3];
  const int*  b2rev = (const int*)d_in[4];
  const int*  mol   = (const int*)d_in[5];
  const float* feats = (const float*)d_in[6];
  const float* Wi = (const float*)d_in[7];
  const float* bi = (const float*)d_in[8];
  const float* Wm = (const float*)d_in[9];
  const float* bm = (const float*)d_in[10];
  const float* Wa = (const float*)d_in[11];
  const float* ba = (const float*)d_in[12];
  float* out = (float*)d_out;

  // workspace layout
  char* base = (char*)d_ws;
  size_t off = 0;
  auto take = [&](size_t bytes) { char* p = base + off; off = (off + bytes + 255) & ~(size_t)255; return p; };
  unsigned short* wi_pk = (unsigned short*)take((size_t)KT0 * NTILES * 512 * 2);
  unsigned short* wm_pk = (unsigned short*)take((size_t)KTM * NTILES * 512 * 2);
  unsigned short* wa_pk = (unsigned short*)take((size_t)KTA * NTILES * 512 * 2);
  float* bi_p = (float*)take(HP * 4);
  float* bm_p = (float*)take(HP * 4);
  float* ba_p = (float*)take(HP * 4);
  unsigned short* h0 = (unsigned short*)take((size_t)EEDGES * HP * 2);  // reused as hv (f32) later
  unsigned short* hA = (unsigned short*)take((size_t)EEDGES * HP * 2);
  unsigned short* hB = (unsigned short*)take((size_t)EEDGES * HP * 2);
  float* inc = (float*)take((size_t)NATOMS * HP * 4);                   // reused as m_v later
  if (off > ws_size) return;  // workspace too small; bail

  // 1) pack weights/biases to WMMA-B tiled bf16
  k_pack_w<<<(KT0 * NTILES * 512 + 255) / 256, 256, 0, stream>>>(Wi, K0R, KT0, wi_pk);
  k_pack_w<<<(KTM * NTILES * 512 + 255) / 256, 256, 0, stream>>>(Wm, HID, KTM, wm_pk);
  k_pack_w<<<(KTA * NTILES * 512 + 255) / 256, 256, 0, stream>>>(Wa, KAR, KTA, wa_pk);
  k_pack_bias<<<2, 256, 0, stream>>>(bi, bi_p);
  k_pack_bias<<<2, 256, 0, stream>>>(bm, bm_p);
  k_pack_bias<<<2, 256, 0, stream>>>(ba, ba_p);

  // 2) h0 GEMM  (E/64 = 3125 exact)
  k_h0<<<EEDGES / 64, 256, 0, stream>>>(x, ea, esrc, wi_pk, bi_p, h0);

  // 3) DEPTH message-passing rounds
  const size_t incN = (size_t)NATOMS * HP;
  const int scatterBlocks = (int)(((size_t)EEDGES * (HP / 2) + 255) / 256);
  const unsigned short* hcur = h0;
  unsigned short* hnxt = hA;
  unsigned short* hoth = hB;
  for (int d = 0; d < 3; ++d) {
    k_zero<<<2048, 256, 0, stream>>>(inc, incN);
    k_scatter<<<scatterBlocks, 256, 0, stream>>>(hcur, edst, inc);
    k_mp<<<EEDGES / 64, 256, 0, stream>>>(esrc, b2rev, inc, hcur, h0, wm_pk, bm_p, hnxt);
    const unsigned short* done = hnxt;
    hnxt = (d == 0) ? hB : (unsigned short*)( (hcur == h0) ? (void*)hA : (void*)hcur );
    // simple explicit ping-pong: after d=0 hcur=hA, after d=1 hcur=hB, after d=2 hcur=hA
    hcur = done;
    hnxt = (hcur == hA) ? hB : hA;
    (void)hoth;
  }
  // hcur now holds final h (hA after 3 rounds)

  // 4) m_v = segment_sum(h, edge_src)   (reuse inc buffer)
  float* mv = inc;
  k_zero<<<2048, 256, 0, stream>>>(mv, incN);
  k_scatter<<<scatterBlocks, 256, 0, stream>>>(hcur, esrc, mv);

  // 5) h_v GEMM (reuse h0 region as f32 [NATOMS][HP])
  float* hv = (float*)h0;
  k_hv<<<(NATOMS + 63) / 64, 256, 0, stream>>>(x, mv, wa_pk, ba_p, hv);

  // 6) readout
  k_outinit<<<(NMOL * OUTW + 255) / 256, 256, 0, stream>>>(out, feats);
  k_mol<<<(int)(((size_t)NATOMS * HID + 255) / 256), 256, 0, stream>>>(hv, mol, out);
}
